// CAInterface_10952166605095
// MI455X (gfx1250) — 8-bit matrix operand order repaired and checked by reading, compiled, not run on hardware
//
#include <hip/hip_runtime.h>
#include <hip/hip_bf16.h>

#define BB   8
#define TT   1024
#define CIN  4096
#define CH   4096
#define MTOT (BB * TT)
#define MB_N (MTOT / 64)
#define NB_N (CH / 64)

typedef __attribute__((ext_vector_type(8))) int v8i;

__global__ void __launch_bounds__(256) ca_pack_int_u8(const int* __restrict__ src,
                                                      unsigned int* __restrict__ dst,
                                                      int n4) {
    int idx = blockIdx.x * 256 + threadIdx.x;
    if (idx >= n4) return;
    const int* s = src + (size_t)idx * 4;
    unsigned int w = 0;
#pragma unroll
    for (int k = 0; k < 4; ++k) w |= (s[k] != 0 ? 1u : 0u) << (8 * k);
    dst[idx] = w;
}

__global__ void __launch_bounds__(256) ca_pack_f32_u8(const float* __restrict__ src,
                                                      unsigned int* __restrict__ dst,
                                                      int n4) {
    int idx = blockIdx.x * 256 + threadIdx.x;
    if (idx >= n4) return;
    const float* s = src + (size_t)idx * 4;
    unsigned int w = 0;
#pragma unroll
    for (int k = 0; k < 4; ++k) w |= (s[k] != 0.0f ? 1u : 0u) << (8 * k);
    dst[idx] = w;
}

__global__ void __launch_bounds__(256) ca_extract_idx(const float* __restrict__ Ah,
                                                      int* __restrict__ idxh) {
    int i = blockIdx.x * 256 + threadIdx.x;
    if (i >= CH) return;
    const float* row = Ah + (size_t)i * CH;
    int c0 = 0, c1 = 0, c2 = 0, cnt = 0;
    for (int j = 0; j < CH; ++j) {
        if (row[j] != 0.0f) {
            if (cnt == 0) c0 = j; else if (cnt == 1) c1 = j; else if (cnt == 2) c2 = j;
            ++cnt;
        }
    }
    idxh[3 * i + 0] = c0;
    idxh[3 * i + 1] = c1;
    idxh[3 * i + 2] = c2;
}

__global__ void __launch_bounds__(256) ca_copy_z(const int* __restrict__ z,
                                                 float* __restrict__ out) {
    size_t idx = (size_t)blockIdx.x * 256 + threadIdx.x;
    size_t m = idx >> 12;
    int    c = (int)(idx & 4095);
    out[m * (CIN + CH) + c] = (z[idx] != 0) ? 1.0f : 0.0f;
}

__global__ void __launch_bounds__(256) ca_wmma_input(const unsigned char* __restrict__ zb,
                                                     const unsigned char* __restrict__ ab,
                                                     unsigned int* __restrict__ inbits) {
    int wave = threadIdx.x >> 5;
    int lane = threadIdx.x & 31;
    int blk  = blockIdx.x * 8 + wave;
    int mb   = blk >> 6;
    int nb   = blk & 63;
    int half = lane >> 4;
    int l16  = lane & 15;

    const unsigned int* zrow[4];
    const unsigned int* arow[4];
#pragma unroll
    for (int r = 0; r < 4; ++r) {
        zrow[r] = (const unsigned int*)(zb + (size_t)(mb * 64 + r * 16 + l16) * CIN);
        arow[r] = (const unsigned int*)(ab + (size_t)(nb * 64 + r * 16 + l16) * CIN);
    }

    v8i acc[4][4];
#pragma unroll
    for (int r = 0; r < 4; ++r)
#pragma unroll
        for (int j = 0; j < 4; ++j)
            acc[r][j] = (v8i){0, 0, 0, 0, 0, 0, 0, 0};

    for (int kk = 0; kk < CIN / 64; ++kk) {
        int kbase = kk << 6;

        v8i bf[4];
#pragma unroll
        for (int j = 0; j < 4; ++j) {
#pragma unroll
            for (int v = 0; v < 8; ++v) {
                int kb = kbase + ((v & 3) << 2) + (half << 4) + ((v >> 2) << 5);
                bf[j][v] = (int)arow[j][kb >> 2];
            }
        }

#pragma unroll
        for (int r = 0; r < 4; ++r) {
            v8i af;
#pragma unroll
            for (int v = 0; v < 8; ++v) {
                int ka = kbase + ((v & 3) << 2) + (half << 4) + ((v >> 2) << 5);
                af[v] = (int)zrow[r][ka >> 2];
            }
#pragma unroll
            for (int j = 0; j < 4; ++j)
                acc[r][j] = __builtin_amdgcn_wmma_i32_16x16x64_iu8(
                    false, af, false, bf[j], acc[r][j], false, false);
        }
    }

#pragma unroll
    for (int r = 0; r < 4; ++r) {
#pragma unroll
        for (int j = 0; j < 4; ++j) {
            unsigned int ball[8];
#pragma unroll
            for (int q = 0; q < 8; ++q)
                ball[q] = __builtin_amdgcn_ballot_w32(acc[r][j][q] > 0);
            if (lane < 16) {
                unsigned int w = ball[lane & 7];
                unsigned short bits = (lane < 8) ? (unsigned short)(w & 0xFFFFu)
                                                 : (unsigned short)(w >> 16);
                int m  = mb * 64 + r * 16 + lane;
                int tn = nb * 4 + j;
                ((unsigned short*)inbits)[(size_t)m * (CH / 16) + tn] = bits;
            }
        }
    }
}

__global__ void __launch_bounds__(1024) ca_scan(const int* __restrict__ h0,
                                                const int* __restrict__ idxh_g,
                                                const unsigned int* __restrict__ inbits,
                                                float* __restrict__ out) {
    __shared__ unsigned int hbuf[2][CH / 32];
    __shared__ int sidx[3 * CH];

    int b = blockIdx.x;
    int tid = threadIdx.x;

    for (int k = tid; k < 3 * CH; k += 1024) sidx[k] = idxh_g[k];

    if (tid < CH / 32) {
        const int* h0row = h0 + (size_t)b * CH + tid * 32;
        unsigned int w = 0;
#pragma unroll
        for (int bit = 0; bit < 32; ++bit) w |= (h0row[bit] != 0 ? 1u : 0u) << bit;
        hbuf[0][tid] = w;
    }
    __syncthreads();

    int wave = tid >> 5, lane = tid & 31;

    for (int t = 0; t < TT; ++t) {
        const unsigned int* cur = hbuf[t & 1];
        unsigned int*       nxt = hbuf[(t + 1) & 1];
        size_t m = (size_t)b * TT + t;
        const unsigned int* inrow = inbits + m * (CH / 32);
        float* orow = out + m * (CIN + CH) + CIN;
#pragma unroll
        for (int it = 0; it < 4; ++it) {
            int w = (wave << 2) + it;
            int i = (w << 5) + lane;
            int j0 = sidx[3 * i + 0];
            int j1 = sidx[3 * i + 1];
            int j2 = sidx[3 * i + 2];
            unsigned int hid = ((cur[j0 >> 5] >> (j0 & 31)) |
                                (cur[j1 >> 5] >> (j1 & 31)) |
                                (cur[j2 >> 5] >> (j2 & 31))) & 1u;
            unsigned int inb = (inrow[w] >> lane) & 1u;
            unsigned int pred = hid & inb;
            unsigned int word = __builtin_amdgcn_ballot_w32(pred != 0);
            if (lane == 0) nxt[w] = word;
            orow[i] = (float)pred;
        }
        __syncthreads();
    }
}

extern "C" void kernel_launch(void* const* d_in, const int* in_sizes, int n_in,
                              void* d_out, int out_size, void* d_ws, size_t ws_size,
                              hipStream_t stream) {
    const int*   z    = (const int*)d_in[0];
    const int*   h0   = (const int*)d_in[1];
    const float* AinF = (const float*)d_in[2];
    const float* AhF  = (const float*)d_in[3];
    float* out = (float*)d_out;

    unsigned char* ws = (unsigned char*)d_ws;
    unsigned char* zb     = ws;
    unsigned char* ab     = ws + (size_t)MTOT * CIN;
    unsigned int*  inbits = (unsigned int*)(ws + 50331648);
    int*           idxh   = (int*)(ws + 54525952);

    ca_pack_int_u8<<<(MTOT * CIN / 4) / 256, 256, 0, stream>>>(z, (unsigned int*)zb,
                                                               MTOT * CIN / 4);
    ca_pack_f32_u8<<<(CH * CIN / 4) / 256, 256, 0, stream>>>(AinF, (unsigned int*)ab,
                                                             CH * CIN / 4);
    ca_extract_idx<<<CH / 256, 256, 0, stream>>>(AhF, idxh);
    ca_copy_z<<<(MTOT * CIN) / 256, 256, 0, stream>>>(z, out);

    ca_wmma_input<<<(MB_N * NB_N) / 8, 256, 0, stream>>>(zb, ab, inbits);

    ca_scan<<<BB, 1024, 0, stream>>>(h0, idxh, inbits, out);
}
